// StandardHORPNHead_32109175505438
// MI455X (gfx1250) — matrix operand order repaired and checked by reading, not compiled, not run on hardware
//
#include <hip/hip_runtime.h>

typedef __attribute__((ext_vector_type(16))) __bf16 v16bf;
typedef __attribute__((ext_vector_type(8)))  float  v8f;

// ---------------- bf16 helpers (bit-exact RNE convert) ----------------
__device__ __forceinline__ __bf16 f2bf(float f) {
    union { float f; unsigned u; } x; x.f = f;
    unsigned r = x.u + 0x7FFFu + ((x.u >> 16) & 1u);
    unsigned short h = (unsigned short)(r >> 16);
    __bf16 b; __builtin_memcpy(&b, &h, 2); return b;
}
__device__ __forceinline__ float bf2f(__bf16 b) {
    unsigned short h; __builtin_memcpy(&h, &b, 2);
    union { unsigned u; float f; } x; x.u = ((unsigned)h) << 16; return x.f;
}

// ---------------- weight packing ----------------
// (OC,256,3,3) f32 -> bf16 [OC][9*256], k = tap*256 + cin  (tap-major K)
__global__ void pack_conv_w(const float* __restrict__ w, __bf16* __restrict__ wp) {
    int i = blockIdx.x * 256 + threadIdx.x;
    if (i >= 256 * 2304) return;
    int k  = i % 2304, oc = i / 2304;
    int tap = k / 256, c = k % 256;
    wp[i] = f2bf(w[((size_t)oc * 256 + c) * 9 + tap]);
}

// generic f32 matrix slice -> bf16 [rows][cols]
__global__ void pack_mat(const float* __restrict__ src, __bf16* __restrict__ dst,
                         int rows, int cols, int srcStride, int srcOff) {
    int i = blockIdx.x * 256 + threadIdx.x;
    if (i >= rows * cols) return;
    int c = i % cols, r = i / cols;
    dst[i] = f2bf(src[(size_t)r * srcStride + srcOff + c]);
}

// NCHW f32 -> [n][S][C] bf16
__global__ void to_nhwc(const float* __restrict__ x, __bf16* __restrict__ y, int S) {
    size_t i = (size_t)blockIdx.x * 256 + threadIdx.x;
    if (i >= (size_t)2 * S * 256) return;
    int c = (int)(i & 255);
    size_t t = i >> 8;
    int pos = (int)(t % S), n = (int)(t / S);
    y[i] = f2bf(x[((size_t)n * 256 + c) * S + pos]);
}

// ---------------- 3x3 conv, implicit GEMM via WMMA bf16 ----------------
// block = 128 threads (4 waves); tile = 32 positions x 64 outC
// each wave: two accumulators (rows 0-15 / 16-31) sharing one B load
// grid = (ceil(S/32), 4, N).  W is a power of two -> shifts, no division.
__global__ __launch_bounds__(128)
void conv3x3_wmma(const __bf16* __restrict__ inT,   // [n][S][256]
                  const __bf16* __restrict__ wPk,   // [256][9*256] tap-major K
                  const float*  __restrict__ bias,  // [256]
                  __bf16* __restrict__ outT,        // [n][S][256], ReLU fused
                  int H, int logW) {
    const int W    = 1 << logW;
    const int S    = H << logW;
    const int n    = blockIdx.z;
    const int p0   = blockIdx.x * 32;
    const int tid  = threadIdx.x;
    const int lane = tid & 31;
    const int wv   = tid >> 5;
    const int g    = lane >> 4;       // K-half select
    const int oc   = blockIdx.y * 64 + wv * 16 + (lane & 15);

    __shared__ __align__(32) __bf16 ldsA[32 * 256];   // 16 KB, one tap, 32 positions

    v8f acc0 = {}, acc1 = {};
    const __bf16* wRow = wPk + (size_t)oc * 2304 + g * 8;    // B: two 8-K chunks 16 K apart per lane
    const char*   aB0  = (const char*)ldsA + (lane & 15) * 512 + g * 16;
    const char*   aB1  = aB0 + 16 * 512;

    for (int tap = 0; tap < 9; ++tap) {
        const int r = tap / 3, s = tap - r * 3;
        __syncthreads();
        // cooperative stage: 32 rows x 512B = 1024 uint4, 128 threads x 8
        #pragma unroll
        for (int i = 0; i < 8; ++i) {
            int idx = tid + i * 128;
            int m = idx >> 5, j = idx & 31;
            int p = p0 + m;
            uint4 val; val.x = val.y = val.z = val.w = 0u;
            if (p < S) {
                int h = p >> logW, w = p & (W - 1);
                int hh = h + r - 1, ww = w + s - 1;
                if (hh >= 0 && hh < H && ww >= 0 && ww < W) {
                    const uint4* src = (const uint4*)(inT + ((size_t)n * S + ((size_t)hh << logW) + ww) * 256);
                    val = src[j];
                }
            }
            ((uint4*)ldsA)[idx] = val;
        }
        __syncthreads();
        const __bf16* wTap = wRow + tap * 256;
        #pragma unroll
        for (int kk = 0; kk < 8; ++kk) {
            union { uint4 q[2]; v16bf v; } bu;           // B reused by both accumulators
            bu.q[0] = *(const uint4*)(wTap + kk * 32);
            bu.q[1] = *(const uint4*)(wTap + kk * 32 + 16);
            v16bf b = bu.v;
            union { uint4 q[2]; v16bf v; } a0, a1;       // A: two 16B chunks 32B apart
            a0.q[0] = *(const uint4*)(aB0 + kk * 64);
            a0.q[1] = *(const uint4*)(aB0 + kk * 64 + 32);
            a1.q[0] = *(const uint4*)(aB1 + kk * 64);
            a1.q[1] = *(const uint4*)(aB1 + kk * 64 + 32);
            acc0 = __builtin_amdgcn_wmma_f32_16x16x32_bf16(false, a0.v, false, b,
                                                           (short)0, acc0, false, false);
            acc1 = __builtin_amdgcn_wmma_f32_16x16x32_bf16(false, a1.v, false, b,
                                                           (short)0, acc1, false, false);
        }
    }
    const float bv = bias[oc];
    #pragma unroll
    for (int v = 0; v < 8; ++v) {
        int m = v + (g << 3);        // C/D layout: lanes>=16 hold M = 8+v
        int p = p0 + m;
        if (p < S)
            outT[((size_t)n * S + p) * 256 + oc] = f2bf(fmaxf(acc0[v] + bv, 0.f));
        int p2 = p + 16;
        if (p2 < S)
            outT[((size_t)n * S + p2) * 256 + oc] = f2bf(fmaxf(acc1[v] + bv, 0.f));
    }
}

// ---------------- generic M x 256 @ 256 x 256 GEMM via WMMA ----------------
// A bf16 [M][256] row-major, Bt bf16 [256][256] = weight[out][in]
// 32-row M tiles, two accumulators per wave sharing the B load.
// out bf16 [M][256], optional bias, optional relu. grid = (ceil(M/32), 4)
template<bool RELU>
__global__ __launch_bounds__(128)
void gemm256_wmma(const __bf16* __restrict__ A, const __bf16* __restrict__ Bt,
                  const float* __restrict__ bias, __bf16* __restrict__ out, int M) {
    const int tid  = threadIdx.x;
    const int lane = tid & 31;
    const int wv   = tid >> 5;
    const int g    = lane >> 4;
    const int m0   = blockIdx.x * 32;
    const int oc   = blockIdx.y * 64 + wv * 16 + (lane & 15);

    int r0 = m0 + (lane & 15);       if (r0 >= M) r0 = M - 1;
    int r1 = m0 + 16 + (lane & 15);  if (r1 >= M) r1 = M - 1;
    const char*   aB0  = (const char*)(A + (size_t)r0 * 256) + g * 16;
    const char*   aB1  = (const char*)(A + (size_t)r1 * 256) + g * 16;
    const __bf16* wRow = Bt + (size_t)oc * 256 + g * 8;

    v8f acc0 = {}, acc1 = {};
    #pragma unroll
    for (int kk = 0; kk < 8; ++kk) {
        union { uint4 q[2]; v16bf v; } bu;
        bu.q[0] = *(const uint4*)(wRow + kk * 32);
        bu.q[1] = *(const uint4*)(wRow + kk * 32 + 16);
        v16bf b = bu.v;
        union { uint4 q[2]; v16bf v; } a0, a1;
        a0.q[0] = *(const uint4*)(aB0 + kk * 64);
        a0.q[1] = *(const uint4*)(aB0 + kk * 64 + 32);
        a1.q[0] = *(const uint4*)(aB1 + kk * 64);
        a1.q[1] = *(const uint4*)(aB1 + kk * 64 + 32);
        acc0 = __builtin_amdgcn_wmma_f32_16x16x32_bf16(false, a0.v, false, b,
                                                       (short)0, acc0, false, false);
        acc1 = __builtin_amdgcn_wmma_f32_16x16x32_bf16(false, a1.v, false, b,
                                                       (short)0, acc1, false, false);
    }
    const float bv = bias ? bias[oc] : 0.f;
    #pragma unroll
    for (int v = 0; v < 8; ++v) {
        int m = m0 + v + (g << 3);
        if (m < M) {
            float f = acc0[v] + bv;
            if (RELU) f = fmaxf(f, 0.f);
            out[(size_t)m * 256 + oc] = f2bf(f);
        }
        int m2 = m + 16;
        if (m2 < M) {
            float f = acc1[v] + bv;
            if (RELU) f = fmaxf(f, 0.f);
            out[(size_t)m2 * 256 + oc] = f2bf(f);
        }
    }
}

// ---------------- 1x1 conv (C=256 -> OC), f32 NCHW output ----------------
__global__ void conv1x1_out(const __bf16* __restrict__ hid, const __bf16* __restrict__ w,
                            const float* __restrict__ b, float* __restrict__ out,
                            int OC, int S) {
    int i = blockIdx.x * 256 + threadIdx.x;
    int total = 2 * OC * S;
    if (i >= total) return;
    int pos = i % S; int t = i / S; int oc = t % OC; int n = t / OC;
    const __bf16* hrow = hid + ((size_t)n * S + pos) * 256;
    const __bf16* wrow = w + (size_t)oc * 256;
    float acc = b[oc];
    #pragma unroll 8
    for (int c = 0; c < 256; ++c) acc += bf2f(hrow[c]) * bf2f(wrow[c]);
    out[i] = acc;   // i == ((n*OC+oc)*S + pos)
}

// ---------------- top-K machinery ----------------
__global__ void cell_scores(const float* __restrict__ plog, float* __restrict__ scores,
                            int S, int off, int Stot) {
    int i = blockIdx.x * 256 + threadIdx.x;
    if (i >= 2 * S) return;
    int n = i / S, pos = i % S;
    float m = plog[((size_t)n * 3 + 0) * S + pos];
    m = fmaxf(m, plog[((size_t)n * 3 + 1) * S + pos]);
    m = fmaxf(m, plog[((size_t)n * 3 + 2) * S + pos]);
    scores[(size_t)n * Stot + off + pos] = m;
}

// global top-20 over all cells + gather pf (f32 [n][20][256]); grid = 2 blocks
__global__ void topk_gather(const float* __restrict__ scores, const __bf16* __restrict__ hidP,
                            float* __restrict__ pf, int S0, int S1, int S2, int S3) {
    const int n = blockIdx.x, tid = threadIdx.x;
    const int Stot = S0 + S1 + S2 + S3;
    __shared__ int   chosen[20];
    __shared__ float sv[256];
    __shared__ int   si[256];
    const float* sc = scores + (size_t)n * Stot;
    for (int t = 0; t < 20; ++t) {
        float best = -1e30f; int bidx = 0;
        for (int i = tid; i < Stot; i += 256) {
            bool used = false;
            for (int u = 0; u < t; ++u) if (chosen[u] == i) used = true;
            float v = sc[i];
            if (!used && v > best) { best = v; bidx = i; }
        }
        sv[tid] = best; si[tid] = bidx;
        __syncthreads();
        for (int off = 128; off > 0; off >>= 1) {
            if (tid < off && sv[tid + off] > sv[tid]) { sv[tid] = sv[tid + off]; si[tid] = si[tid + off]; }
            __syncthreads();
        }
        if (tid == 0) chosen[t] = si[0];
        __syncthreads();
    }
    for (int e = tid; e < 20 * 256; e += 256) {
        int k = e >> 8, c = e & 255;
        int idx = chosen[k];
        int pos, Sl; size_t base;
        if (idx < S0)               { pos = idx;                 Sl = S0; base = 0; }
        else if (idx < S0 + S1)     { pos = idx - S0;            Sl = S1; base = (size_t)2 * S0 * 256; }
        else if (idx < S0 + S1 + S2){ pos = idx - S0 - S1;       Sl = S2; base = (size_t)2 * (S0 + S1) * 256; }
        else                        { pos = idx - S0 - S1 - S2;  Sl = S3; base = (size_t)2 * (S0 + S1 + S2) * 256; }
        pf[((size_t)n * 20 + k) * 256 + c] = bf2f(hidP[base + ((size_t)n * Sl + pos) * 256 + c]);
    }
}

// p_proj[n][k][d] = pf[n][k][:] . Wp[d][:] + rn1_b[d]
__global__ void pproj_kernel(const float* __restrict__ pf, const __bf16* __restrict__ Wp,
                             const float* __restrict__ b, float* __restrict__ pp) {
    int i = blockIdx.x * 256 + threadIdx.x;
    if (i >= 2 * 20 * 256) return;
    int d = i & 255, nk = i >> 8;
    const float* row = pf + (size_t)nk * 256;
    const __bf16* w = Wp + (size_t)d * 256;
    float acc = b[d];
    #pragma unroll 8
    for (int c = 0; c < 256; ++c) acc += row[c] * bf2f(w[c]);
    pp[i] = acc;
}

// h[n][s][d] = sum_k relu(o_proj + p_proj[n][k][d]) -> bf16
__global__ void hsum_kernel(const __bf16* __restrict__ oproj, const float* __restrict__ pp,
                            __bf16* __restrict__ hB, int S) {
    size_t i = (size_t)blockIdx.x * 256 + threadIdx.x;
    if (i >= (size_t)2 * S * 256) return;
    int d = (int)(i & 255);
    size_t row = i >> 8;
    int n = (int)(row / S);
    float o = bf2f(oproj[i]);
    const float* p = pp + (size_t)n * 20 * 256 + d;
    float s = 0.f;
    #pragma unroll
    for (int k = 0; k < 20; ++k) s += fmaxf(o + p[k * 256], 0.f);
    hB[i] = f2bf(s);
}

// ==================== host launch ====================
extern "C" void kernel_launch(void* const* d_in, const int* in_sizes, int n_in,
                              void* d_out, int out_size, void* d_ws, size_t ws_size,
                              hipStream_t stream) {
    (void)in_sizes; (void)n_in; (void)out_size; (void)ws_size;
    static const int SL[4]   = {10240, 2560, 640, 160};
    static const int HL[4]   = {80, 40, 20, 10};
    static const int LOGW[4] = {7, 6, 5, 4};
    static const int CUM[4]  = {0, 10240, 12800, 13440};
    const int STOT = 13600;

    const float* feat[4] = {(const float*)d_in[0], (const float*)d_in[1],
                            (const float*)d_in[2], (const float*)d_in[3]};
    const float* pc1_w = (const float*)d_in[4];  const float* pc1_b = (const float*)d_in[5];
    const float* pc2_w = (const float*)d_in[6];  const float* pc2_b = (const float*)d_in[7];
    const float* oc1_w = (const float*)d_in[8];  const float* oc1_b = (const float*)d_in[9];
    const float* oc2_w = (const float*)d_in[10]; const float* oc2_b = (const float*)d_in[11];
    const float* rn1_w = (const float*)d_in[12]; const float* rn1_b = (const float*)d_in[13];
    const float* rn2_w = (const float*)d_in[14]; const float* rn2_b = (const float*)d_in[15];
    const float* pl_w  = (const float*)d_in[16]; const float* pl_b  = (const float*)d_in[17];
    const float* pd_w  = (const float*)d_in[18]; const float* pd_b  = (const float*)d_in[19];
    const float* ol_w  = (const float*)d_in[20]; const float* ol_b  = (const float*)d_in[21];
    const float* od_w  = (const float*)d_in[22]; const float* od_b  = (const float*)d_in[23];

    float* out = (float*)d_out;
    // output offsets (floats): p_log x4, p_del x4, o_log x4, o_del x4
    size_t offPlog[4], offPdel[4], offOlog[4], offOdel[4];
    for (int l = 0; l < 4; ++l) {
        offPlog[l] = (size_t)6  * CUM[l];
        offPdel[l] = 81600  + (size_t)24 * CUM[l];
        offOlog[l] = 408000 + (size_t)6  * CUM[l];
        offOdel[l] = 489600 + (size_t)24 * CUM[l];
    }

    // ---- workspace carve (256B aligned) ----
    char* cur = (char*)d_ws;
    auto alloc = [&](size_t bytes) -> char* {
        char* r = cur; cur += (bytes + 255) & ~(size_t)255; return r;
    };
    __bf16* WPC1 = (__bf16*)alloc((size_t)256 * 2304 * 2);
    __bf16* WPC2 = (__bf16*)alloc((size_t)256 * 2304 * 2);
    __bf16* WOC1 = (__bf16*)alloc((size_t)256 * 2304 * 2);
    __bf16* WOC2 = (__bf16*)alloc((size_t)256 * 2304 * 2);
    __bf16* WPL  = (__bf16*)alloc((size_t)3  * 256 * 2);
    __bf16* WPD  = (__bf16*)alloc((size_t)12 * 256 * 2);
    __bf16* WOD  = (__bf16*)alloc((size_t)12 * 256 * 2);
    __bf16* WOL  = (__bf16*)alloc((size_t)3  * 256 * 2);
    __bf16* WWO  = (__bf16*)alloc((size_t)256 * 256 * 2);
    __bf16* WWP  = (__bf16*)alloc((size_t)256 * 256 * 2);
    __bf16* WRN2 = (__bf16*)alloc((size_t)256 * 256 * 2);
    __bf16* FT   = (__bf16*)alloc((size_t)2 * STOT * 256 * 2);   // all levels, NHWC bf16
    __bf16* HP   = (__bf16*)alloc((size_t)2 * STOT * 256 * 2);   // person hidden, all levels
    __bf16* TMP  = (__bf16*)alloc((size_t)2 * SL[0] * 256 * 2);  // conv1 scratch (max level)
    __bf16* HO   = (__bf16*)alloc((size_t)2 * SL[0] * 256 * 2);  // object hidden (per level)
    __bf16* OPRJ = (__bf16*)alloc((size_t)2 * SL[0] * 256 * 2);  // o_proj (per level)
    __bf16* HB   = (__bf16*)alloc((size_t)2 * SL[0] * 256 * 2);  // h (per level)
    __bf16* H2   = (__bf16*)alloc((size_t)2 * SL[0] * 256 * 2);  // relu(h@rn2) (per level)
    float*  SCOR = (float*)alloc((size_t)2 * STOT * 4);
    float*  PF   = (float*)alloc((size_t)2 * 20 * 256 * 4);
    float*  PPRJ = (float*)alloc((size_t)2 * 20 * 256 * 4);

    // ---- pack weights to bf16 ----
    pack_conv_w<<<2304, 256, 0, stream>>>(pc1_w, WPC1);
    pack_conv_w<<<2304, 256, 0, stream>>>(pc2_w, WPC2);
    pack_conv_w<<<2304, 256, 0, stream>>>(oc1_w, WOC1);
    pack_conv_w<<<2304, 256, 0, stream>>>(oc2_w, WOC2);
    pack_mat<<<3,   256, 0, stream>>>(pl_w,  WPL,  3,   256, 256, 0);
    pack_mat<<<12,  256, 0, stream>>>(pd_w,  WPD,  12,  256, 256, 0);
    pack_mat<<<12,  256, 0, stream>>>(od_w,  WOD,  12,  256, 256, 0);
    pack_mat<<<3,   256, 0, stream>>>(ol_w,  WOL,  3,   256, 256, 0);
    pack_mat<<<256, 256, 0, stream>>>(rn1_w, WWO,  256, 256, 512, 0);    // Wo = rn1_w[:, :256]
    pack_mat<<<256, 256, 0, stream>>>(rn1_w, WWP,  256, 256, 512, 256);  // Wp = rn1_w[:, 256:]
    pack_mat<<<256, 256, 0, stream>>>(rn2_w, WRN2, 256, 256, 256, 0);

    // ---- person branch per level ----
    for (int l = 0; l < 4; ++l) {
        const int S = SL[l];
        __bf16* FTl = FT + (size_t)2 * CUM[l] * 256;
        __bf16* HPl = HP + (size_t)2 * CUM[l] * 256;
        int nel = 2 * S * 256;
        to_nhwc<<<(nel + 255) / 256, 256, 0, stream>>>(feat[l], FTl, S);
        dim3 cg((S + 31) / 32, 4, 2);
        conv3x3_wmma<<<cg, 128, 0, stream>>>(FTl, WPC1, pc1_b, TMP, HL[l], LOGW[l]);
        conv3x3_wmma<<<cg, 128, 0, stream>>>(TMP, WPC2, pc2_b, HPl, HL[l], LOGW[l]);
        conv1x1_out<<<(2 * 3  * S + 255) / 256, 256, 0, stream>>>(HPl, WPL, pl_b, out + offPlog[l], 3,  S);
        conv1x1_out<<<(2 * 12 * S + 255) / 256, 256, 0, stream>>>(HPl, WPD, pd_b, out + offPdel[l], 12, S);
    }

    // ---- top-K + p_proj ----
    for (int l = 0; l < 4; ++l)
        cell_scores<<<(2 * SL[l] + 255) / 256, 256, 0, stream>>>(out + offPlog[l], SCOR, SL[l], CUM[l], STOT);
    topk_gather<<<2, 256, 0, stream>>>(SCOR, HP, PF, SL[0], SL[1], SL[2], SL[3]);
    pproj_kernel<<<(2 * 20 * 256 + 255) / 256, 256, 0, stream>>>(PF, WWP, rn1_b, PPRJ);

    // ---- object branch + relational per level ----
    for (int l = 0; l < 4; ++l) {
        const int S = SL[l];
        const int M = 2 * S;
        __bf16* FTl = FT + (size_t)2 * CUM[l] * 256;
        dim3 cg((S + 31) / 32, 4, 2);
        conv3x3_wmma<<<cg, 128, 0, stream>>>(FTl, WOC1, oc1_b, TMP, HL[l], LOGW[l]);
        conv3x3_wmma<<<cg, 128, 0, stream>>>(TMP, WOC2, oc2_b, HO, HL[l], LOGW[l]);
        conv1x1_out<<<(2 * 12 * S + 255) / 256, 256, 0, stream>>>(HO, WOD, od_b, out + offOdel[l], 12, S);
        dim3 gg((M + 31) / 32, 4);
        gemm256_wmma<false><<<gg, 128, 0, stream>>>(HO, WWO, nullptr, OPRJ, M);       // o_proj
        hsum_kernel<<<((size_t)M * 256 + 255) / 256, 256, 0, stream>>>(OPRJ, PPRJ, HB, S);
        gemm256_wmma<true><<<gg, 128, 0, stream>>>(HB, WRN2, rn2_b, H2, M);           // relu(h@rn2+b)
        conv1x1_out<<<(2 * 3 * S + 255) / 256, 256, 0, stream>>>(H2, WOL, ol_b, out + offOlog[l], 3, S);
    }
}
